// node_update_26517128085911
// MI455X (gfx1250) — matrix operand order repaired and checked by reading, not compiled, not run on hardware
//
#include <hip/hip_runtime.h>
#include <float.h>

// ============================================================================
// MI455X (gfx1250, wave32) implementation.
// Heavy ops = dense GEMMs -> v_wmma_f32_16x16x32_bf16 with fp32 accumulate.
// fp32 activations/weights converted to bf16 while staging into LDS.
// ============================================================================

typedef __bf16 bf16;
typedef __bf16 v16bf __attribute__((ext_vector_type(16)));
typedef float  v8f   __attribute__((ext_vector_type(8)));

#define LDT 72  // LDS row stride in bf16 elems (64 + 8 pad -> no bank conflicts)

// ---- WMMA fragment loads (contiguous k order, same in A and B; wave32) -------------
// A (16x32 bf16): lane holds row M=lane&15; half=lane>>4;
//   elems 0..15 -> K = half*16 + e
__device__ __forceinline__ v16bf frag_a(const bf16* lds, int row0, int sOff, int lane) {
  const bf16* p = lds + (row0 + (lane & 15)) * LDT + sOff + (lane >> 4) * 16;
  v16bf f;
#pragma unroll
  for (int e = 0; e < 8; ++e) f[e] = p[e];
#pragma unroll
  for (int e = 0; e < 8; ++e) f[8 + e] = p[8 + e];
  return f;
}
// B (32x16 bf16, stored in LDS as [n][k]): lane holds column N=lane&15;
//   elems 0..15 -> K = (lane>>4)*16 + e   (row-per-VGPR layout, mirrors C/D)
__device__ __forceinline__ v16bf frag_b(const bf16* lds, int col0, int sOff, int lane) {
  const bf16* p = lds + (col0 + (lane & 15)) * LDT + sOff + (lane >> 4) * 16;
  v16bf f;
#pragma unroll
  for (int e = 0; e < 16; ++e) f[e] = p[e];
  return f;
}

// ---- Generic GEMM:  C[R x Co] = concat(A-segments)[R x Kd] @ W[Kd x Co] + bias
// Segment modes: 0 = direct row, 1 = broadcast over K=4 users (row>>2),
//                2 = leave-one-out max over the K=4 user group.
// Block tile 64x64, KC=64, 256 threads = 8 waves; wave grid 4(M) x 2(N),
// each wave -> 16x32 of C (two 16x16 WMMA accumulators).
__global__ __launch_bounds__(256)
void gemm_bf16(float* __restrict__ C, int ldc,
               const float* __restrict__ bias, int relu,
               int Rr, int Co, int Kd,
               const float* a0, int lda0, int w0, int m0,
               const float* a1, int lda1, int w1, int m1,
               const float* a2, int lda2, int w2, int m2,
               const float* __restrict__ W) {
  __shared__ bf16 lA[64 * LDT];
  __shared__ bf16 lB[64 * LDT];
  const int tid  = threadIdx.x;
  const int lane = tid & 31;
  const int wave = tid >> 5;
  const int wm = wave >> 1;              // 0..3 : 16-row band
  const int wn = wave & 1;               // 0..1 : 32-col band
  const int rowBase = blockIdx.y * 64;
  const int colBase = blockIdx.x * 64;
  const int srow = tid >> 2;             // staging row (A) / column (B), 0..63
  const int skc  = (tid & 3) << 4;       // staging k offset: 0/16/32/48

  v8f acc0, acc1;
#pragma unroll
  for (int i = 0; i < 8; ++i) { acc0[i] = 0.f; acc1[i] = 0.f; }

  for (int k0 = 0; k0 < Kd; k0 += 64) {
    __syncthreads();
    // ---- stage A tile (64 rows x 64 k) fp32 -> bf16, segment-aware --------
    {
      float vals[16];
#pragma unroll
      for (int e = 0; e < 16; ++e) vals[e] = 0.f;
      const int grow = rowBase + srow;
      const int kk = k0 + skc;
      if (grow < Rr && kk < Kd) {
        // 16-chunks never straddle segments in our calls (widths % 512 == 0
        // for multi-segment GEMMs; single-segment otherwise).
        const float* p = a0; int lda = lda0, mode = m0, width = w0, kloc = kk;
        if (kloc >= w0) {
          kloc -= w0; p = a1; lda = lda1; mode = m1; width = w1;
          if (kloc >= w1) { kloc -= w1; p = a2; lda = lda2; mode = m2; width = w2; }
        }
        if (mode == 2) {                 // leave-one-out max over users
          const int bb = grow >> 2, self = grow & 3;
          const float* r0 = p + (size_t)(bb * 4) * lda + kloc;
#pragma unroll
          for (int e = 0; e < 16; ++e) {
            if (kloc + e < width) {
              float mx = -FLT_MAX;
#pragma unroll
              for (int j = 0; j < 4; ++j)
                if (j != self) mx = fmaxf(mx, r0[(size_t)j * lda + e]);
              vals[e] = mx;
            }
          }
        } else {
          const int r = (mode == 1) ? (grow >> 2) : grow;  // mode1: ris broadcast
          const float* q = p + (size_t)r * lda + kloc;
          if (kloc + 16 <= width) {
#pragma unroll
            for (int e = 0; e < 16; ++e) vals[e] = q[e];
          } else {
#pragma unroll
            for (int e = 0; e < 16; ++e) if (kloc + e < width) vals[e] = q[e];
          }
        }
      }
#pragma unroll
      for (int e = 0; e < 16; ++e) lA[srow * LDT + skc + e] = (bf16)vals[e];
    }
    // ---- stage B tile (64 k x 64 n) transposed into [n][k], fp32 -> bf16 --
    {
      const int gn = colBase + srow;
#pragma unroll
      for (int e = 0; e < 16; ++e) {
        const int gk = k0 + skc + e;
        float v = 0.f;
        if (gn < Co && gk < Kd) v = W[(size_t)gk * Co + gn];
        lB[srow * LDT + skc + e] = (bf16)v;
      }
      if (gn < Co && k0 + 64 < Kd)       // gfx1250 global_prefetch_b8
        __builtin_prefetch(&W[(size_t)(k0 + 64 + skc) * Co + gn], 0, 1);
    }
    __syncthreads();
    // ---- compute: 2 k-steps x (1 A-frag, 2 B-frags, 2 WMMAs) --------------
#pragma unroll
    for (int s = 0; s < 2; ++s) {
      v16bf fa  = frag_a(lA, wm * 16, s * 32, lane);
      v16bf fb0 = frag_b(lB, wn * 32,      s * 32, lane);
      v16bf fb1 = frag_b(lB, wn * 32 + 16, s * 32, lane);
      acc0 = __builtin_amdgcn_wmma_f32_16x16x32_bf16(false, fa, false, fb0,
                                                     (short)0, acc0, false, false);
      acc1 = __builtin_amdgcn_wmma_f32_16x16x32_bf16(false, fa, false, fb1,
                                                     (short)0, acc1, false, false);
    }
  }
  // ---- epilogue: bias + optional relu; C/D layout: N=lane&15, M=v+8*(lane>>4)
  const int n  = colBase + wn * 32 + (lane & 15);
  const int mB = rowBase + wm * 16 + (lane >> 4) * 8;
#pragma unroll
  for (int v = 0; v < 8; ++v) {
    const int m = mB + v;
    if (m < Rr) {
      if (n < Co) {
        float x = acc0[v] + bias[n];
        if (relu) x = fmaxf(x, 0.f);
        C[(size_t)m * ldc + n] = x;
      }
      if (n + 16 < Co) {
        float x = acc1[v] + bias[n + 16];
        if (relu) x = fmaxf(x, 0.f);
        C[(size_t)m * ldc + n + 16] = x;
      }
    }
  }
}

// ============================ elementwise kernels ============================
// X1[b*4+k] = [user_node(b,k,:) (1024) | cbu moveaxis (16)]
__global__ void pack_x1(float* X, const float* un, const float* cbu, int total) {
  int i = blockIdx.x * blockDim.x + threadIdx.x; if (i >= total) return;
  int r = i / 1040, c = i % 1040, b = r >> 2, k = r & 3;
  float v;
  if (c < 1024) v = un[(size_t)r * 1024 + c];
  else { int cc = c - 1024; v = cbu[(size_t)b * 64 + (cc >> 3) * 32 + (cc & 7) * 4 + k]; }
  X[i] = v;
}
// Xc rows = (b, h*4+w) over M=8,K=4 -> 32; cols = [RIS_node(128) | cbu(2)]
__global__ void pack_xc(float* X, const float* ris, const float* cbu, int total) {
  int i = blockIdx.x * blockDim.x + threadIdx.x; if (i >= total) return;
  int r = i / 130, c = i % 130, b = r >> 5, hw = r & 31;
  X[i] = (c < 128) ? ris[(size_t)b * 4096 + c * 32 + hw]
                   : cbu[(size_t)b * 64 + (c - 128) * 32 + hw];
}
__global__ void pack_x2(float* X, const float* un1, const float* cru, int total) {
  int i = blockIdx.x * blockDim.x + threadIdx.x; if (i >= total) return;
  int r = i / 520, c = i % 520, b = r >> 2, k = r & 3;
  float v;
  if (c < 512) v = un1[(size_t)r * 512 + c];
  else { int cc = c - 512; v = cru[(size_t)b * 32 + (cc >> 2) * 16 + (cc & 3) * 4 + k]; }
  X[i] = v;
}
__global__ void pack_xc2(float* X, const float* ris1, const float* cru, int total) {
  int i = blockIdx.x * blockDim.x + threadIdx.x; if (i >= total) return;
  int r = i / 130, c = i % 130, b = r >> 4, hw = r & 15;
  X[i] = (c < 128) ? ris1[(size_t)b * 2048 + c * 16 + hw]
                   : cru[(size_t)b * 32 + (c - 128) * 16 + hw];
}
// out[b, colOff+c] = mean_g in[b*G+g, c]
__global__ void mean_rows(float* out, int ldo, int colOff, const float* in,
                          int ldi, int ncols, int G, int total) {
  int i = blockIdx.x * blockDim.x + threadIdx.x; if (i >= total) return;
  int b = i / ncols, c = i % ncols;
  float s = 0.f;
  for (int g = 0; g < G; ++g) s += in[(size_t)(b * G + g) * ldi + c];
  out[(size_t)b * ldo + colOff + c] = s / (float)G;
}
// meanU[b, c] = mean over K=4 users of U[b*4+k, c]
__global__ void mean_k4(float* out, const float* U, int ldu, int f, int total) {
  int i = blockIdx.x * blockDim.x + threadIdx.x; if (i >= total) return;
  int b = i / f, c = i % f;
  const float* p = U + (size_t)(b * 4) * ldu + c;
  out[(size_t)b * f + c] = 0.25f * (p[0] + p[(size_t)ldu] + p[(size_t)2 * ldu] + p[(size_t)3 * ldu]);
}
__global__ void copy_cols(float* dst, int ldd, int dcol, const float* src,
                          int lsrc, int scol, int ncols, int total) {
  int i = blockIdx.x * blockDim.x + threadIdx.x; if (i >= total) return;
  int r = i / ncols, c = i % ncols;
  dst[(size_t)r * ldd + dcol + c] = src[(size_t)r * lsrc + scol + c];
}
// in: (B*4, C) -> out: (B, C, 4) normalized over K with scale (sqrt(PT)/K)
__global__ void normalize_k4(float* out, const float* in, int C, float scale, int total) {
  int i = blockIdx.x * blockDim.x + threadIdx.x; if (i >= total) return;
  int b = i / C, c = i % C;
  float v0 = in[(size_t)(b * 4 + 0) * C + c], v1 = in[(size_t)(b * 4 + 1) * C + c];
  float v2 = in[(size_t)(b * 4 + 2) * C + c], v3 = in[(size_t)(b * 4 + 3) * C + c];
  float nr = fmaxf(sqrtf(v0 * v0 + v1 * v1 + v2 * v2 + v3 * v3), 1e-12f);
  float s = scale / nr;
  float* o = out + ((size_t)b * C + c) * 4;
  o[0] = v0 * s; o[1] = v1 * s; o[2] = v2 * s; o[3] = v3 * s;
}
// o: (B,128) -> Theta (B,64,2), unit-normalized pairs (o[n], o[64+n])
__global__ void theta_norm(float* out, const float* o, int total) {
  int i = blockIdx.x * blockDim.x + threadIdx.x; if (i >= total) return;
  int b = i / 64, n = i % 64;
  float a = o[(size_t)b * 128 + n], c = o[(size_t)b * 128 + 64 + n];
  float nr = fmaxf(sqrtf(a * a + c * c), 1e-12f);
  out[(size_t)b * 128 + n * 2]     = a / nr;
  out[(size_t)b * 128 + n * 2 + 1] = c / nr;
}

// ================================ host side =================================
struct Seg { const float* p; int lda; int w; int mode; };

static inline void gemm(hipStream_t st, float* C, int ldc, const float* bias,
                        int relu, int Rr, int Co, Seg s0, Seg s1, Seg s2,
                        const float* W) {
  const int Kd = s0.w + s1.w + s2.w;
  dim3 g((Co + 63) / 64, (Rr + 63) / 64);
  gemm_bf16<<<g, 256, 0, st>>>(C, ldc, bias, relu, Rr, Co, Kd,
                               s0.p, s0.lda, s0.w, s0.mode,
                               s1.p, s1.lda, s1.w, s1.mode,
                               s2.p, s2.lda, s2.w, s2.mode, W);
}
#define EW(kern, n, ...) kern<<<((n) + 255) / 256, 256, 0, stream>>>(__VA_ARGS__, (n))

extern "C" void kernel_launch(void* const* d_in, const int* in_sizes, int n_in,
                              void* d_out, int out_size, void* d_ws, size_t ws_size,
                              hipStream_t stream) {
  (void)in_sizes; (void)n_in; (void)out_size; (void)ws_size;
  const float* RIS_node      = (const float*)d_in[0];
  const float* user_node     = (const float*)d_in[1];
  const float* RIS_node_1    = (const float*)d_in[2];
  const float* user_node_1   = (const float*)d_in[3];
  const float* cru           = (const float*)d_in[6]; // channel_relay_user
  const float* cbu           = (const float*)d_in[7]; // channel_bs_user
  const float* relay_channel = (const float*)d_in[8];
  auto P = [&](int i) { return (const float*)d_in[i]; };
  // params pytree leaves (dict keys sorted, lists/tuples in order):
  //  9..12 init_RIS_C(w1,b1,w2,b2)  13..16 init_RIS_FE  17..20 init_user_C
  //  21..44 layers[d]{f2:(w1,b1,w2,b2), f3:(w1,b1,w2,b2)}  45..48 p2_RIS_C
  //  49..72 p2_layers  73..76 p2_user_C  77..78 readout_BS  79..80 readout_RIS
  //  81..82 readout_RIS2  83..84 readout_relay

  float* ws   = (float*)d_ws;
  float* U    = ws;                           // 4096 x 2048 (phase1 user, grows)
  float* U2   = U    + (size_t)4096 * 2048;   // 4096 x 4096 (phase2 user)
  float* R    = U2   + (size_t)4096 * 4096;   // 1024 x 2048 (phase1 ris)
  float* R2   = R    + (size_t)1024 * 2048;   // 1024 x 4096 (phase2 ris)
  float* H    = R2   + (size_t)1024 * 4096;   // hidden scratch (max 32768x512)
  float* X    = H    + (size_t)32768 * 512;   // packed inputs (max 4096x1040)
  float* CC   = X    + (size_t)4096 * 1040;   // conv outputs (max 8.39M)
  float* MEAN = CC   + (size_t)32768 * 256;   // mean-over-users (max 1024x3584)
  float* WB   = MEAN + (size_t)1024 * 3584;   // 4096 x 16
  float* RB   = WB   + (size_t)4096 * 16;     // 4096 x 8
  float* O    = RB   + (size_t)4096 * 8;      // 1024 x 128

  float* out     = (float*)d_out;
  float* oW      = out;             // (B,16,4)  65536
  float* oRelay  = out + 65536;     // (B, 8,4)  32768
  float* oTheta  = out + 98304;     // (B,64,2) 131072
  float* oTheta2 = out + 229376;    // (B,64,2) 131072

  const Seg Z{nullptr, 1, 0, 0};

  // ================= Phase 1 =================
  EW(pack_x1, 4096 * 1040, X, user_node, cbu);
  gemm(stream, H, 1024, P(18), 1, 4096, 1024, Seg{X, 1040, 1040, 0}, Z, Z, P(17));
  gemm(stream, U, 2048, P(20), 0, 4096, 512, Seg{H, 1024, 1024, 0}, Z, Z, P(19));

  EW(pack_xc, 32768 * 130, X, RIS_node, cbu);
  gemm(stream, H, 512, P(10), 1, 32768, 512, Seg{X, 130, 130, 0}, Z, Z, P(9));
  gemm(stream, CC, 256, P(12), 0, 32768, 256, Seg{H, 512, 512, 0}, Z, Z, P(11));
  EW(mean_rows, 1024 * 256, R, 2048, 0, CC, 256, 256, 32);
  gemm(stream, H, 512, P(14), 1, 1024, 512, Seg{relay_channel, 64, 64, 0}, Z, Z, P(13));
  gemm(stream, R + 256, 2048, P(16), 0, 1024, 256, Seg{H, 512, 512, 0}, Z, Z, P(15));

  for (int d = 0; d < 3; ++d) {
    const int f = 512 * (d + 1);
    const int pb = 21 + d * 8;
    EW(mean_k4, 1024 * f, MEAN, U, 2048, f);
    gemm(stream, H, 512, P(pb + 1), 1, 4096, 512,
         Seg{U, 2048, f, 0}, Seg{U, 2048, f, 2}, Seg{R, 2048, f, 1}, P(pb));
    gemm(stream, U + f, 2048, P(pb + 3), 0, 4096, 512, Seg{H, 512, 512, 0}, Z, Z, P(pb + 2));
    gemm(stream, H, 512, P(pb + 5), 1, 1024, 512,
         Seg{R, 2048, f, 0}, Seg{MEAN, f, f, 0}, Z, P(pb + 4));
    gemm(stream, R + f, 2048, P(pb + 7), 0, 1024, 512, Seg{H, 512, 512, 0}, Z, Z, P(pb + 6));
  }
  gemm(stream, WB, 16, P(78), 0, 4096, 16, Seg{U, 2048, 2048, 0}, Z, Z, P(77));
  EW(normalize_k4, 1024 * 16, oW, WB, 16, 0.25f);
  gemm(stream, O, 128, P(80), 0, 1024, 128, Seg{R, 2048, 2048, 0}, Z, Z, P(79));
  EW(theta_norm, 1024 * 64, oTheta, O);

  // ================= Phase 2 =================
  EW(pack_x2, 4096 * 520, X, user_node_1, cru);
  gemm(stream, H, 1024, P(74), 1, 4096, 1024, Seg{X, 520, 520, 0}, Z, Z, P(73));
  gemm(stream, U2, 4096, P(76), 0, 4096, 512, Seg{H, 1024, 1024, 0}, Z, Z, P(75));
  EW(copy_cols, 4096 * 2048, U2, 4096, 512, U, 2048, 0, 2048);

  EW(pack_xc2, 16384 * 130, X, RIS_node_1, cru);
  gemm(stream, H, 512, P(46), 1, 16384, 512, Seg{X, 130, 130, 0}, Z, Z, P(45));
  gemm(stream, CC, 512, P(48), 0, 16384, 512, Seg{H, 512, 512, 0}, Z, Z, P(47));
  EW(mean_rows, 1024 * 512, R2, 4096, 0, CC, 512, 512, 16);
  EW(copy_cols, 1024 * 2048, R2, 4096, 512, R, 2048, 0, 2048);

  for (int d = 0; d < 3; ++d) {
    const int f = 2560 + 512 * d;
    const int pb = 49 + d * 8;
    EW(mean_k4, 1024 * f, MEAN, U2, 4096, f);
    gemm(stream, H, 512, P(pb + 1), 1, 4096, 512,
         Seg{U2, 4096, f, 0}, Seg{U2, 4096, f, 2}, Seg{R2, 4096, f, 1}, P(pb));
    gemm(stream, U2 + f, 4096, P(pb + 3), 0, 4096, 512, Seg{H, 512, 512, 0}, Z, Z, P(pb + 2));
    gemm(stream, H, 512, P(pb + 5), 1, 1024, 512,
         Seg{R2, 4096, f, 0}, Seg{MEAN, f, f, 0}, Z, P(pb + 4));
    gemm(stream, R2 + f, 4096, P(pb + 7), 0, 1024, 512, Seg{H, 512, 512, 0}, Z, Z, P(pb + 6));
  }
  gemm(stream, RB, 8, P(84), 0, 4096, 8, Seg{U2, 4096, 4096, 0}, Z, Z, P(83));
  EW(normalize_k4, 1024 * 8, oRelay, RB, 8, 0.25f);
  gemm(stream, O, 128, P(82), 0, 1024, 128, Seg{R2, 4096, 4096, 0}, Z, Z, P(81));
  EW(theta_norm, 1024 * 64, oTheta2, O);
}
